// Hybrid_GNN_MLP_87608742904192
// MI455X (gfx1250) — matrix operand order repaired and checked by reading, not compiled, not run on hardware
//
#include <hip/hip_runtime.h>

#define NUM_USERS 100000
#define NUM_ITEMS 50000
#define N_NODES   150000
#define D_EMB     64
#define NNZ_E     2400000
#define BATCH     4096
#define HIST_L    50
#define UHIST_L   30
#define MC_L      5
#define CONCAT_D  192
#define INNER_D   384

typedef __attribute__((ext_vector_type(16))) _Float16 v16h;
typedef __attribute__((ext_vector_type(8)))  float    v8f;

// ---------------------------------------------------------------------------
// SpMM: e_dst[row] += val * e_src[col], one wave (32 lanes) per edge,
// 2 floats per lane. L2-resident table (38.4MB < 192MB L2).
// ---------------------------------------------------------------------------
__global__ void spmm_kernel(const float* __restrict__ esrc,
                            float* __restrict__ edst,
                            const int* __restrict__ rows,
                            const int* __restrict__ cols,
                            const float* __restrict__ vals,
                            int nnz) {
    long long tid  = (long long)blockIdx.x * blockDim.x + threadIdx.x;
    long long edge = tid >> 5;
    int lane = (int)(tid & 31);
    if (edge >= nnz) return;
    int r = rows[edge];
    int c = cols[edge];
    float v = vals[edge];
    // prefetch a future edge's source row (gfx1250 global_prefetch_b8)
    long long pe = edge + 256;
    if (pe < nnz) {
        __builtin_prefetch(&esrc[(long long)cols[pe] * D_EMB + 2 * lane], 0, 1);
    }
    const float2 s = *(const float2*)(esrc + (long long)c * D_EMB + 2 * lane);
    float* dst = edst + (long long)r * D_EMB + 2 * lane;
    atomicAdd(dst,     v * s.x);
    atomicAdd(dst + 1, v * s.y);
}

__global__ void accadd_kernel(float* __restrict__ acc,
                              const float* __restrict__ e, int n) {
    int i = blockIdx.x * blockDim.x + threadIdx.x;
    if (i < n) acc[i] += e[i];
}

// ---------------------------------------------------------------------------
// Convert MLP weights to f16 AND transpose to N-major so each WMMA lane's
// B-fragment (16 consecutive K at fixed N) is one contiguous 32-byte load.
//   w1t[blk][n][k] = w1[blk][k][n]   (k<192, n<384)
//   w2t[blk][n][k] = w2[blk][k][n]   (k<384, n<192)
// ---------------------------------------------------------------------------
__global__ void cvt_kernel(const float* __restrict__ w1,
                           const float* __restrict__ w2,
                           _Float16* __restrict__ w1t,
                           _Float16* __restrict__ w2t, int n) {
    int i = blockIdx.x * blockDim.x + threadIdx.x;
    if (i >= n) return;
    {
        int blk = i / (CONCAT_D * INNER_D);
        int rem = i % (CONCAT_D * INNER_D);
        int k   = rem / INNER_D;
        int c   = rem % INNER_D;
        w1t[((size_t)blk * INNER_D + c) * CONCAT_D + k] = (_Float16)w1[i];
    }
    {
        int blk = i / (INNER_D * CONCAT_D);
        int rem = i % (INNER_D * CONCAT_D);
        int k   = rem / CONCAT_D;
        int c   = rem % CONCAT_D;
        w2t[((size_t)blk * CONCAT_D + c) * INNER_D + k] = (_Float16)w2[i];
    }
}

// ---------------------------------------------------------------------------
// Feature assembly: one block per batch row, one lane per embedding dim.
// ---------------------------------------------------------------------------
__global__ void feature_kernel(const float* __restrict__ acc,     // GNN acc (pre /4)
                               const float* __restrict__ emb,
                               const float* __restrict__ cate_table,
                               const int* __restrict__ cates,
                               const int* __restrict__ cate_lens,
                               const int* __restrict__ users,
                               const int* __restrict__ items,
                               const int* __restrict__ ihm,
                               const int* __restrict__ ihl,
                               const int* __restrict__ uhm,
                               const int* __restrict__ uhl,
                               float* __restrict__ uf,
                               float* __restrict__ itf) {
    int b = blockIdx.x;
    int d = threadIdx.x;            // 0..63
    // ---- user features ----
    int u = users[b];
    uf[b * CONCAT_D + d] = 0.25f * acc[(long long)u * D_EMB + d];
    int ilen = ihl[b];
    float s_item = 0.f, s_cate = 0.f;
    for (int h = 0; h < ilen && h < HIST_L; ++h) {
        int it = ihm[b * HIST_L + h];
        s_item += emb[(long long)(it + NUM_USERS) * D_EMB + d];
        int cl = cate_lens[it];
        float sc = 0.f;
        for (int c = 0; c < cl && c < MC_L; ++c)
            sc += cate_table[(long long)cates[it * MC_L + c] * D_EMB + d];
        s_cate += sc / ((float)cl + 1e-9f);
    }
    float inv = 1.0f / ((float)ilen + 1e-9f);
    uf[b * CONCAT_D + D_EMB + d]     = s_item * inv;
    uf[b * CONCAT_D + 2 * D_EMB + d] = s_cate * inv;
    // ---- item features ----
    int it0 = items[b];
    itf[b * CONCAT_D + d] = 0.25f * acc[(long long)(it0 + NUM_USERS) * D_EMB + d];
    int cl0 = cate_lens[it0];
    float sc0 = 0.f;
    for (int c = 0; c < cl0 && c < MC_L; ++c)
        sc0 += cate_table[(long long)cates[it0 * MC_L + c] * D_EMB + d];
    itf[b * CONCAT_D + D_EMB + d] = sc0 / ((float)cl0 + 1e-9f);
    int ulen = uhl[b];
    float sh = 0.f;
    for (int h = 0; h < ulen && h < UHIST_L; ++h)
        sh += emb[(long long)uhm[b * UHIST_L + h] * D_EMB + d];
    itf[b * CONCAT_D + 2 * D_EMB + d] = sh / ((float)ulen + 1e-9f);
}

// ---------------------------------------------------------------------------
// MLP: 16 batch rows per workgroup, 128 threads = 4 waves (wave32).
// Two PreNormResidual blocks summed, then row L2-normalize.
// GEMMs via v_wmma_f32_16x16x32_f16 (f16 in, f32 acc).
// B fragments: one aligned v16h (2x global_load_b128) from N-major weights.
// blockIdx.y: 0 = user path (MLP blocks 0,1 -> ue), 1 = item path (2,3 -> ie).
// ---------------------------------------------------------------------------
__global__ __launch_bounds__(128) void mlp_kernel(
        const float* __restrict__ uf, const float* __restrict__ itf,
        const float* __restrict__ lnw, const float* __restrict__ lnb,
        const float* __restrict__ b1,  const float* __restrict__ b2,
        const _Float16* __restrict__ w1t, const _Float16* __restrict__ w2t,
        float* __restrict__ out) {
    __shared__ float    xs[16][200];     // input tile f32
    __shared__ _Float16 ys[16][200];     // LN output f16
    __shared__ _Float16 hs[16][392];     // hidden f16
    __shared__ float    os[16][200];     // accumulator for the two blocks
    __shared__ float    mu_s[16], rs_s[16], inv_s[16];

    const int tid  = threadIdx.x;
    const int wave = tid >> 5;
    const int lane = tid & 31;
    const int m    = lane & 15;          // N (for B/C/D) or M (for A) lane index
    const int half = lane >> 4;
    const int row0 = blockIdx.x * 16;
    const float* feat = (blockIdx.y == 0) ? uf : itf;
    float* outp = out + (size_t)blockIdx.y * BATCH * CONCAT_D;
    const int blk0 = blockIdx.y * 2;

    // load input tile + zero accumulator
    for (int t = tid; t < 16 * CONCAT_D; t += 128) {
        int r = t / CONCAT_D, c = t % CONCAT_D;
        xs[r][c] = feat[(size_t)(row0 + r) * CONCAT_D + c];
        os[r][c] = 0.f;
    }
    __syncthreads();

    for (int p = 0; p < 2; ++p) {
        const int blk = blk0 + p;
        // ---- LayerNorm ----
        if (tid < 16) {
            float s = 0.f, s2 = 0.f;
            for (int c = 0; c < CONCAT_D; ++c) { float v = xs[tid][c]; s += v; s2 += v * v; }
            float mu  = s / (float)CONCAT_D;
            float var = s2 / (float)CONCAT_D - mu * mu;
            mu_s[tid] = mu;
            rs_s[tid] = rsqrtf(var + 1e-5f);
        }
        __syncthreads();
        for (int t = tid; t < 16 * CONCAT_D; t += 128) {
            int r = t / CONCAT_D, c = t % CONCAT_D;
            ys[r][c] = (_Float16)((xs[r][c] - mu_s[r]) * rs_s[r] *
                                  lnw[blk * CONCAT_D + c] + lnb[blk * CONCAT_D + c]);
        }
        __syncthreads();

        // ---- GEMM1: h = relu(y @ W1 + b1), [16,192]x[192,384] ----
        for (int nt = wave; nt < INNER_D / 16; nt += 4) {
            v8f c = {};
            const int ncol = nt * 16 + m;
            const _Float16* wcol = w1t + ((size_t)blk * INNER_D + ncol) * CONCAT_D;
            for (int ks = 0; ks < CONCAT_D / 32; ++ks) {
                const int kb0 = ks * 32;
                v16h a;
#pragma unroll
                for (int v = 0; v < 8; ++v) {
                    int ka = kb0 + 2 * v + 16 * half;
                    a[2 * v]     = ys[m][ka];
                    a[2 * v + 1] = ys[m][ka + 1];
                }
                v16h bf = *(const v16h*)(wcol + kb0 + 16 * half);
                c = __builtin_amdgcn_wmma_f32_16x16x32_f16(false, a, false, bf,
                                                           (short)0, c, false, false);
            }
            const float bias = b1[blk * INNER_D + ncol];
#pragma unroll
            for (int r = 0; r < 8; ++r) {
                float v = c[r] + bias;
                hs[r + 8 * half][ncol] = (_Float16)(v > 0.f ? v : 0.f);
            }
        }
        __syncthreads();

        // ---- GEMM2: os += h @ W2 + b2 + x, [16,384]x[384,192] ----
        for (int nt = wave; nt < CONCAT_D / 16; nt += 4) {
            v8f c = {};
            const int ncol = nt * 16 + m;
            const _Float16* wcol = w2t + ((size_t)blk * CONCAT_D + ncol) * INNER_D;
            for (int ks = 0; ks < INNER_D / 32; ++ks) {
                const int kb0 = ks * 32;
                v16h a;
#pragma unroll
                for (int v = 0; v < 8; ++v) {
                    int ka = kb0 + 2 * v + 16 * half;
                    a[2 * v]     = hs[m][ka];
                    a[2 * v + 1] = hs[m][ka + 1];
                }
                v16h bf = *(const v16h*)(wcol + kb0 + 16 * half);
                c = __builtin_amdgcn_wmma_f32_16x16x32_f16(false, a, false, bf,
                                                           (short)0, c, false, false);
            }
            const float bias = b2[blk * CONCAT_D + ncol];
#pragma unroll
            for (int r = 0; r < 8; ++r) {
                int mm = r + 8 * half;
                os[mm][ncol] += c[r] + bias + xs[mm][ncol];
            }
        }
        __syncthreads();
    }

    // ---- row L2 normalize ----
    if (tid < 16) {
        float ss = 0.f;
        for (int c = 0; c < CONCAT_D; ++c) { float v = os[tid][c]; ss += v * v; }
        float nrm = sqrtf(ss);
        inv_s[tid] = 1.0f / fmaxf(nrm, 1e-12f);
    }
    __syncthreads();
    for (int t = tid; t < 16 * CONCAT_D; t += 128) {
        int r = t / CONCAT_D, c = t % CONCAT_D;
        outp[(size_t)(row0 + r) * CONCAT_D + c] = os[r][c] * inv_s[r];
    }
}

// ---------------------------------------------------------------------------
extern "C" void kernel_launch(void* const* d_in, const int* in_sizes, int n_in,
                              void* d_out, int out_size, void* d_ws, size_t ws_size,
                              hipStream_t stream) {
    const float* emb        = (const float*)d_in[0];
    const float* cate_table = (const float*)d_in[1];
    const float* adj_vals   = (const float*)d_in[2];
    const float* mlp_ln_w   = (const float*)d_in[3];
    const float* mlp_ln_b   = (const float*)d_in[4];
    const float* mlp_w1     = (const float*)d_in[5];
    const float* mlp_b1     = (const float*)d_in[6];
    const float* mlp_w2     = (const float*)d_in[7];
    const float* mlp_b2     = (const float*)d_in[8];
    const int*   adj_rows   = (const int*)d_in[9];
    const int*   adj_cols   = (const int*)d_in[10];
    const int*   cates      = (const int*)d_in[11];
    const int*   cate_lens  = (const int*)d_in[12];
    const int*   users      = (const int*)d_in[13];
    const int*   items      = (const int*)d_in[14];
    const int*   ihm        = (const int*)d_in[15];
    const int*   ihl        = (const int*)d_in[16];
    const int*   uhm        = (const int*)d_in[17];
    const int*   uhl        = (const int*)d_in[18];

    const size_t n_e    = (size_t)N_NODES * D_EMB;           // 9.6M floats
    const size_t ebytes = n_e * sizeof(float);
    const size_t n_w    = 4 * CONCAT_D * INNER_D;            // 294912 (same for W1, W2)

    char* ws   = (char*)d_ws;
    float*    acc  = (float*)ws;                 ws += ebytes;
    float*    buf0 = (float*)ws;                 ws += ebytes;
    float*    buf1 = (float*)ws;                 ws += ebytes;
    float*    uf   = (float*)ws;                 ws += (size_t)BATCH * CONCAT_D * sizeof(float);
    float*    itf  = (float*)ws;                 ws += (size_t)BATCH * CONCAT_D * sizeof(float);
    _Float16* w1t  = (_Float16*)ws;              ws += n_w * sizeof(_Float16);
    _Float16* w2t  = (_Float16*)ws;

    // acc = e0
    hipMemcpyAsync(acc, emb, ebytes, hipMemcpyDeviceToDevice, stream);

    const int spmm_blocks = (int)(((long long)NNZ_E * 32 + 255) / 256);
    const int acc_blocks  = (int)((n_e + 255) / 256);

    // layer 1: e1 = A@e0
    hipMemsetAsync(buf0, 0, ebytes, stream);
    spmm_kernel<<<spmm_blocks, 256, 0, stream>>>(emb, buf0, adj_rows, adj_cols, adj_vals, NNZ_E);
    accadd_kernel<<<acc_blocks, 256, 0, stream>>>(acc, buf0, (int)n_e);
    // layer 2: e2 = A@e1
    hipMemsetAsync(buf1, 0, ebytes, stream);
    spmm_kernel<<<spmm_blocks, 256, 0, stream>>>(buf0, buf1, adj_rows, adj_cols, adj_vals, NNZ_E);
    accadd_kernel<<<acc_blocks, 256, 0, stream>>>(acc, buf1, (int)n_e);
    // layer 3: e3 = A@e2  (buf0's e1 no longer needed -> reuse)
    hipMemsetAsync(buf0, 0, ebytes, stream);
    spmm_kernel<<<spmm_blocks, 256, 0, stream>>>(buf1, buf0, adj_rows, adj_cols, adj_vals, NNZ_E);
    accadd_kernel<<<acc_blocks, 256, 0, stream>>>(acc, buf0, (int)n_e);

    // weights -> f16, transposed to N-major for contiguous B fragments
    cvt_kernel<<<(int)((n_w + 255) / 256), 256, 0, stream>>>(mlp_w1, mlp_w2, w1t, w2t, (int)n_w);

    // feature assembly (gnn = acc/4 folded in)
    feature_kernel<<<BATCH, 64, 0, stream>>>(acc, emb, cate_table, cates, cate_lens,
                                             users, items, ihm, ihl, uhm, uhl, uf, itf);

    // WMMA MLP: y=0 -> ue, y=1 -> ie
    mlp_kernel<<<dim3(BATCH / 16, 2), 128, 0, stream>>>(uf, itf, mlp_ln_w, mlp_ln_b,
                                                        mlp_b1, mlp_b2, w1t, w2t,
                                                        (float*)d_out);
}
